// NonLocalBlock2D_54107997995618
// MI455X (gfx1250) — matrix operand order repaired and checked by reading, not compiled, not run on hardware
//
#include <hip/hip_runtime.h>
#include <hip/hip_bf16.h>
#include <math.h>

#define B_  8
#define C_  128
#define CI_ 64
#define N_  4096

typedef _Float16 h16v __attribute__((ext_vector_type(16)));
typedef _Float16 h8v  __attribute__((ext_vector_type(8)));
typedef float    f8v  __attribute__((ext_vector_type(8)));

static __device__ __forceinline__ h16v cat8(h8v lo, h8v hi) {
  return __builtin_shufflevector(lo, hi, 0,1,2,3,4,5,6,7,8,9,10,11,12,13,14,15);
}
static __device__ __forceinline__ f8v wmma_f16(h16v a, h16v b, f8v c) {
  // D = A(16x32 f16) * B(32x16 f16) + C(16x16 f32)
  return __builtin_amdgcn_wmma_f32_16x16x32_f16(false, a, false, b, (short)0, c,
                                                false, false);
}
// reduce over the 16 lanes of each wave32 half (xor 1/2/4/8 stays in-half)
static __device__ __forceinline__ float redmax16(float v) {
  v = fmaxf(v, __shfl_xor(v, 1, 32));
  v = fmaxf(v, __shfl_xor(v, 2, 32));
  v = fmaxf(v, __shfl_xor(v, 4, 32));
  v = fmaxf(v, __shfl_xor(v, 8, 32));
  return v;
}
static __device__ __forceinline__ float redsum16(float v) {
  v += __shfl_xor(v, 1, 32);
  v += __shfl_xor(v, 2, 32);
  v += __shfl_xor(v, 4, 32);
  v += __shfl_xor(v, 8, 32);
  return v;
}

// ---------------------------------------------------------------------------
// Stage A: 1x1 conv projections in fp32, emit f16 copies laid out for WMMA.
//   thetaH[b][n][ci]  (contraction dim ci contiguous -> B-tile rows)
//   phiH  [b][m][ci]  (== phi^T, rows of A tiles)
//   gT    [b][ci][m]  (m contiguous -> B-tile rows for the P^T*g GEMM)
// ---------------------------------------------------------------------------
__global__ __launch_bounds__(256) void nlb_proj(
    const float* __restrict__ x,
    const float* __restrict__ g_w,  const float* __restrict__ g_b,
    const float* __restrict__ th_w, const float* __restrict__ th_b,
    const float* __restrict__ ph_w, const float* __restrict__ ph_b,
    _Float16* __restrict__ thetaH, _Float16* __restrict__ phiH,
    _Float16* __restrict__ gT)
{
  __shared__ float xt[C_ * 33];
  const int tid = threadIdx.x;
  const int bb  = blockIdx.x >> 7;           // batch
  const int n0  = (blockIdx.x & 127) * 32;   // 32 spatial positions per block
  for (int idx = tid; idx < C_ * 32; idx += 256) {
    int c = idx >> 5, p = idx & 31;
    xt[c * 33 + p] = x[((size_t)(bb * C_ + c)) * N_ + n0 + p];
  }
  __syncthreads();
  for (int it = tid; it < 3 * CI_ * 32; it += 256) {
    int proj = it >> 11;          // 0=g, 1=theta, 2=phi (uniform per iteration)
    int rem  = it & 2047;
    int ci   = rem >> 5;
    int p    = rem & 31;
    const float* w; float acc;
    if (proj == 0)      { w = g_w  + ci * C_; acc = g_b[ci];  }
    else if (proj == 1) { w = th_w + ci * C_; acc = th_b[ci]; }
    else                { w = ph_w + ci * C_; acc = ph_b[ci]; }
    #pragma unroll 8
    for (int c = 0; c < C_; ++c) acc += w[c] * xt[c * 33 + p];
    if (proj == 0)
      gT[((size_t)(bb * CI_ + ci)) * N_ + n0 + p] = (_Float16)acc;
    else if (proj == 1)
      thetaH[((size_t)(bb * N_ + n0 + p)) * CI_ + ci] = (_Float16)acc;
    else
      phiH[((size_t)(bb * N_ + n0 + p)) * CI_ + ci] = (_Float16)acc;
  }
}

// ---------------------------------------------------------------------------
// Stage B: per-m softmax stats over all n -> statML[m] = max + log(sumexp).
// One wave32 per (b, 32-row m-pair-tile). Per-LANE online max/sum (each lane
// owns its own n subset); single cross-lane combine at the end.
// ---------------------------------------------------------------------------
__global__ __launch_bounds__(32) void nlb_stats(
    const _Float16* __restrict__ thetaH, const _Float16* __restrict__ phiH,
    float* __restrict__ statML)
{
  const int lane = threadIdx.x;
  const int l15  = lane & 15;
  const int kh   = lane >> 4;
  const int bb   = blockIdx.x >> 7;
  const int m0   = (blockIdx.x & 127) * 32;   // two 16-row m-tiles

  // A tiles (phi^T rows, K = ci): loop-invariant, 2 m-tiles x 2 k-halves
  h16v a[2][2];
  #pragma unroll
  for (int mt = 0; mt < 2; ++mt) {
    const _Float16* arow = phiH + ((size_t)(bb * N_ + m0 + mt * 16 + l15)) * CI_;
    a[mt][0] = cat8(*(const h8v*)(arow + kh * 16),      *(const h8v*)(arow + kh * 16 + 8));
    a[mt][1] = cat8(*(const h8v*)(arow + 32 + kh * 16), *(const h8v*)(arow + 32 + kh * 16 + 8));
  }

  float runM[2][8], runL[2][8];
  #pragma unroll
  for (int mt = 0; mt < 2; ++mt)
    #pragma unroll
    for (int r = 0; r < 8; ++r) { runM[mt][r] = -__builtin_inff(); runL[mt][r] = 0.f; }

  const _Float16* thbase = thetaH + (size_t)bb * N_ * CI_;
  for (int n0 = 0; n0 < N_; n0 += 64) {
    f8v acc[2][4];
    #pragma unroll
    for (int j = 0; j < 4; ++j) {
      const _Float16* brow = thbase + ((size_t)(n0 + 16 * j + l15)) * CI_;
      h16v b0 = cat8(*(const h8v*)(brow + kh * 16),      *(const h8v*)(brow + kh * 16 + 8));
      h16v b1 = cat8(*(const h8v*)(brow + 32 + kh * 16), *(const h8v*)(brow + 32 + kh * 16 + 8));
      #pragma unroll
      for (int mt = 0; mt < 2; ++mt) {
        f8v c = {};
        c = wmma_f16(a[mt][0], b0, c);
        c = wmma_f16(a[mt][1], b1, c);
        acc[mt][j] = c;
      }
    }
    // per-lane online update (no cross-lane traffic in the hot loop)
    #pragma unroll
    for (int mt = 0; mt < 2; ++mt)
      #pragma unroll
      for (int r = 0; r < 8; ++r) {
        float lm = fmaxf(fmaxf(acc[mt][0][r], acc[mt][1][r]),
                         fmaxf(acc[mt][2][r], acc[mt][3][r]));
        float nm = fmaxf(runM[mt][r], lm);
        float es = __expf(acc[mt][0][r] - nm) + __expf(acc[mt][1][r] - nm)
                 + __expf(acc[mt][2][r] - nm) + __expf(acc[mt][3][r] - nm);
        runL[mt][r] = runL[mt][r] * __expf(runM[mt][r] - nm) + es;
        runM[mt][r] = nm;
      }
  }
  // final cross-lane combine: ML = gmax + log(sum)
  #pragma unroll
  for (int mt = 0; mt < 2; ++mt)
    #pragma unroll
    for (int r = 0; r < 8; ++r) {
      float gm = redmax16(runM[mt][r]);
      float ls = runL[mt][r] * __expf(runM[mt][r] - gm);
      float tot = redsum16(ls);
      if (l15 == 0)
        statML[bb * N_ + m0 + mt * 16 + r + 8 * kh] = gm + __logf(tot);
    }
}

// ---------------------------------------------------------------------------
// Stage C: y[n,:] = sum_m exp(S[m,n]-ML[m]) * g[m,:].
// One wave32 per (b, 32-col n-pair-tile). Recompute S tiles, exp, transpose P
// through LDS into A-layout, then accumulate P^T (16n x 32m) * g (32m x 16ci).
// phi/g tiles are shared across the two n-tiles.
// ---------------------------------------------------------------------------
__global__ __launch_bounds__(32) void nlb_attn(
    const _Float16* __restrict__ thetaH, const _Float16* __restrict__ phiH,
    const _Float16* __restrict__ gT, const float* __restrict__ statML,
    float* __restrict__ y)
{
  __shared__ _Float16 PT[32 * 32];   // PT[n_local][m_local]
  __shared__ float MLs[32];
  const int lane = threadIdx.x;
  const int l15  = lane & 15;
  const int kh   = lane >> 4;
  const int bb   = blockIdx.x >> 7;
  const int n0   = (blockIdx.x & 127) * 32;   // two 16-col n-tiles

  // B tiles for S (theta rows): loop-invariant, 2 n-tiles x 2 k-halves
  h16v bt[2][2];
  #pragma unroll
  for (int nt = 0; nt < 2; ++nt) {
    const _Float16* brow = thetaH + ((size_t)(bb * N_ + n0 + nt * 16 + l15)) * CI_;
    bt[nt][0] = cat8(*(const h8v*)(brow + kh * 16),      *(const h8v*)(brow + kh * 16 + 8));
    bt[nt][1] = cat8(*(const h8v*)(brow + 32 + kh * 16), *(const h8v*)(brow + 32 + kh * 16 + 8));
  }

  const float* MLb = statML + bb * N_;
  const _Float16* phb = phiH + (size_t)bb * N_ * CI_;
  const _Float16* gtb = gT   + (size_t)bb * CI_ * N_;

  f8v zero = {};
  f8v acc[2][4];
  #pragma unroll
  for (int nt = 0; nt < 2; ++nt)
    #pragma unroll
    for (int ct = 0; ct < 4; ++ct) acc[nt][ct] = zero;

  for (int m0 = 0; m0 < N_; m0 += 32) {
    MLs[lane] = MLb[m0 + lane];
    __syncthreads();
    #pragma unroll
    for (int t = 0; t < 2; ++t) {
      const _Float16* arow = phb + ((size_t)(m0 + 16 * t + l15)) * CI_;
      h16v a0 = cat8(*(const h8v*)(arow + kh * 16),      *(const h8v*)(arow + kh * 16 + 8));
      h16v a1 = cat8(*(const h8v*)(arow + 32 + kh * 16), *(const h8v*)(arow + 32 + kh * 16 + 8));
      #pragma unroll
      for (int nt = 0; nt < 2; ++nt) {
        f8v s = zero;
        s = wmma_f16(a0, bt[nt][0], s);
        s = wmma_f16(a1, bt[nt][1], s);
        #pragma unroll
        for (int r = 0; r < 8; ++r) {
          float p = __expf(s[r] - MLs[16 * t + r + 8 * kh]);
          PT[(nt * 16 + l15) * 32 + 16 * t + r + 8 * kh] = (_Float16)p;
        }
      }
    }
    __syncthreads();
    // A' = P^T tiles (rows = n, K = 32 m's): contiguous 16B LDS loads
    h16v ap[2];
    #pragma unroll
    for (int nt = 0; nt < 2; ++nt)
      ap[nt] = cat8(*(const h8v*)&PT[(nt * 16 + l15) * 32 + kh * 16],
                    *(const h8v*)&PT[(nt * 16 + l15) * 32 + kh * 16 + 8]);
    #pragma unroll
    for (int ct = 0; ct < 4; ++ct) {
      const _Float16* grow = gtb + ((size_t)(ct * 16 + l15)) * N_ + m0 + kh * 16;
      h16v bg = cat8(*(const h8v*)grow, *(const h8v*)(grow + 8));
      #pragma unroll
      for (int nt = 0; nt < 2; ++nt)
        acc[nt][ct] = wmma_f16(ap[nt], bg, acc[nt][ct]);
    }
    __syncthreads();
  }
  #pragma unroll
  for (int nt = 0; nt < 2; ++nt)
    #pragma unroll
    for (int ct = 0; ct < 4; ++ct)
      #pragma unroll
      for (int r = 0; r < 8; ++r)
        y[((size_t)(bb * N_ + n0 + nt * 16 + r + 8 * kh)) * CI_ + ct * 16 + l15] =
            acc[nt][ct][r];
}

// ---------------------------------------------------------------------------
// Stage D: out = W_w @ y + W_b + x  (fp32, LDS-staged y tile)
// ---------------------------------------------------------------------------
__global__ __launch_bounds__(256) void nlb_out(
    const float* __restrict__ x, const float* __restrict__ W_w,
    const float* __restrict__ W_b, const float* __restrict__ y,
    float* __restrict__ out)
{
  __shared__ float yt[64 * 65];
  const int tid = threadIdx.x;
  const int bb  = blockIdx.x >> 6;
  const int n0  = (blockIdx.x & 63) * 64;
  for (int idx = tid; idx < 64 * 64; idx += 256) {
    int p = idx >> 6, ci = idx & 63;
    yt[p * 65 + ci] = y[((size_t)(bb * N_ + n0 + p)) * CI_ + ci];
  }
  __syncthreads();
  for (int it = tid; it < C_ * 64; it += 256) {
    int co = it >> 6;
    int p  = it & 63;
    const float* w = W_w + co * CI_;
    float acc = W_b[co];
    #pragma unroll 8
    for (int ci = 0; ci < CI_; ++ci) acc += w[ci] * yt[p * 65 + ci];
    size_t oi = ((size_t)(bb * C_ + co)) * N_ + n0 + p;
    out[oi] = acc + x[oi];
  }
}

extern "C" void kernel_launch(void* const* d_in, const int* in_sizes, int n_in,
                              void* d_out, int out_size, void* d_ws, size_t ws_size,
                              hipStream_t stream) {
  const float* x    = (const float*)d_in[0];
  const float* g_w  = (const float*)d_in[1];
  const float* g_b  = (const float*)d_in[2];
  const float* th_w = (const float*)d_in[3];
  const float* th_b = (const float*)d_in[4];
  const float* ph_w = (const float*)d_in[5];
  const float* ph_b = (const float*)d_in[6];
  const float* W_w  = (const float*)d_in[7];
  const float* W_b  = (const float*)d_in[8];
  float* out = (float*)d_out;

  char* ws = (char*)d_ws;
  size_t off = 0;
  auto carve = [&](size_t bytes) -> void* {
    void* p = ws + off;
    off = (off + bytes + 255) & ~(size_t)255;
    return p;
  };
  _Float16* thetaH = (_Float16*)carve((size_t)B_ * N_ * CI_ * sizeof(_Float16));
  _Float16* phiH   = (_Float16*)carve((size_t)B_ * N_ * CI_ * sizeof(_Float16));
  _Float16* gT     = (_Float16*)carve((size_t)B_ * CI_ * N_ * sizeof(_Float16));
  float*    statML = (float*)carve((size_t)B_ * N_ * sizeof(float));
  float*    yb     = (float*)carve((size_t)B_ * N_ * CI_ * sizeof(float));

  nlb_proj<<<dim3(B_ * (N_ / 32)), dim3(256), 0, stream>>>(
      x, g_w, g_b, th_w, th_b, ph_w, ph_b, thetaH, phiH, gT);
  nlb_stats<<<dim3(B_ * (N_ / 32)), dim3(32), 0, stream>>>(
      thetaH, phiH, statML);
  nlb_attn<<<dim3(B_ * (N_ / 32)), dim3(32), 0, stream>>>(
      thetaH, phiH, gT, statML, yb);
  nlb_out<<<dim3(B_ * (N_ / 64)), dim3(256), 0, stream>>>(
      x, W_w, W_b, yb, out);
}
